// Prediction_layer_27127013441815
// MI455X (gfx1250) — matrix operand order repaired and checked by reading, not compiled, not run on hardware
//
#include <hip/hip_runtime.h>
#include <hip/hip_bf16.h>

typedef __attribute__((ext_vector_type(16))) __bf16 v16bf;
typedef __attribute__((ext_vector_type(8)))  float  v8f;

#define N_ROWS 2048
#define BATCH  64
#define D_IN   768
#define D_MID  512
#define D_OUT  128

#define TILE_M 16
#define KT1 (D_IN / 32)    // 24 K-tiles for layer 1
#define KT2 (D_MID / 32)   // 16 K-tiles for layer 2
#define FRAG_ELEMS 512     // one 16x32 (or 32x16) bf16 fragment = 512 elems

// workspace layout (bytes)
#define W1P_BYTES ((size_t)D_IN * D_MID * 2)    // 786432
#define W2P_BYTES ((size_t)D_MID * D_OUT * 2)   // 131072
#define W1P_OFF   ((size_t)0)
#define W2P_OFF   (W1P_OFF + W1P_BYTES)
#define ACC_OFF   (W2P_OFF + W2P_BYTES)         // BATCH*D_OUT u32

// order-preserving float <-> u32 encoding so u32 atomic max == float max
__device__ __forceinline__ unsigned enc_f32(float f) {
  unsigned u = __float_as_uint(f);
  return (u & 0x80000000u) ? ~u : (u | 0x80000000u);
}
__device__ __forceinline__ float dec_f32(unsigned u) {
  return __uint_as_float((u & 0x80000000u) ? (u ^ 0x80000000u) : ~u);
}

// A-matrix 16x32 bf16 fragment mapping (ISA 7.12.2):
// lanes 0-15: v0-3 -> K=0..7, v4-7 -> K=16..23 ; lanes 16-31: v0-3 -> K=8..15, v4-7 -> K=24..31
__device__ __forceinline__ void a_map(int m, int kl, int& lane, int& e) {
  if (kl < 8)       { lane = m;      e = kl;      }
  else if (kl < 16) { lane = m + 16; e = kl - 8;  }
  else if (kl < 24) { lane = m;      e = kl - 8;  }
  else              { lane = m + 16; e = kl - 16; }
}

// Repack fp32 row-major W (K x Ntot) into bf16 B-fragment layout:
// fragment (tk, tn) stored contiguously at ((tn*ktiles + tk) * 512),
// within a fragment: lane*16 + e, lanes 0-15 hold K=0..7,16..23, lanes 16-31 hold K=8..15,24..31.
__global__ void pack_w(const float* __restrict__ W, __bf16* __restrict__ out,
                       int K, int Ntot) {
  int total = K * Ntot;
  int ktiles = K >> 5;
  for (int i = blockIdx.x * blockDim.x + threadIdx.x; i < total;
       i += gridDim.x * blockDim.x) {
    int k = i / Ntot, n = i - k * Ntot;
    int tk = k >> 5, kl = k & 31;
    int tn = n >> 4, col = n & 15;
    int lane, e;
    lane = col + 16 * ((kl >> 3) & 1);
    e = (kl & 7) + 8 * (kl >> 4);
    out[((size_t)(tn * ktiles + tk)) * FRAG_ELEMS + lane * 16 + e] = (__bf16)W[i];
  }
}

__global__ void init_acc(unsigned* __restrict__ acc) {
  int i = blockIdx.x * blockDim.x + threadIdx.x;
  if (i < BATCH * D_OUT) acc[i] = enc_f32(-INFINITY);
}

// One workgroup = one (batch b, 16-row tile). 256 threads = 8 wave32s.
__global__ __launch_bounds__(256) void fused_mlp(
    const float* __restrict__ x, const float* __restrict__ b1,
    const float* __restrict__ b2, const int* __restrict__ batch_list,
    const __bf16* __restrict__ w1p, const __bf16* __restrict__ w2p,
    unsigned* __restrict__ acc) {
  const int b    = blockIdx.y;
  const int row0 = blockIdx.x * TILE_M;
  const int start = (b == 0) ? 0 : batch_list[b - 1];
  const int len   = batch_list[b];
  int end = start + len;
  if (end > N_ROWS) end = N_ROWS;
  if (row0 >= end || row0 + TILE_M <= start) return;  // tile outside segment

  __shared__ __align__(32) __bf16 aF[KT1 * FRAG_ELEMS];  // 24 KB: x tile, A-frag layout
  __shared__ __align__(32) __bf16 hF[KT2 * FRAG_ELEMS];  // 16 KB: h tile, A-frag layout

  const int tid = threadIdx.x;

  // Stage x tile (16 rows x 768) -> bf16 A-fragment layout in LDS
  for (int i = tid; i < TILE_M * D_IN; i += 256) {
    int m = i / D_IN;
    int k = i - m * D_IN;
    float v = x[(size_t)(row0 + m) * (BATCH * D_IN) + (size_t)b * D_IN + k];
    int tk = k >> 5, kl = k & 31;
    int lane, e;
    a_map(m, kl, lane, e);
    aF[tk * FRAG_ELEMS + lane * 16 + e] = (__bf16)v;
  }
  __syncthreads();

  const int wave = tid >> 5;
  const int lane = tid & 31;
  const v8f vzero = {};

  // ---- Layer 1: h = relu(x @ W1 + b1); wave w owns cols [w*64, w*64+64) ----
  v8f c1[4];
#pragma unroll
  for (int j = 0; j < 4; ++j) c1[j] = vzero;
  for (int tk = 0; tk < KT1; ++tk) {
    v16bf a = *reinterpret_cast<const v16bf*>(&aF[tk * FRAG_ELEMS + lane * 16]);
#pragma unroll
    for (int j = 0; j < 4; ++j) {
      const v16bf bm = *reinterpret_cast<const v16bf*>(
          &w1p[((size_t)(wave * 4 + j) * KT1 + tk) * FRAG_ELEMS + lane * 16]);
      c1[j] = __builtin_amdgcn_wmma_f32_16x16x32_bf16(false, a, false, bm,
                                                      (short)0, c1[j], false, false);
    }
  }
  // bias + relu, scatter into h A-frag layout (C/D layout: col = lane%16, row = r + 8*(lane>=16))
  {
    const int mBase = (lane >= 16) ? 8 : 0;
    const int ncol  = lane & 15;
#pragma unroll
    for (int j = 0; j < 4; ++j) {
      int n0 = wave * 64 + j * 16 + ncol;
      float bias = b1[n0];
      int tk2 = n0 >> 5, kl2 = n0 & 31;
#pragma unroll
      for (int r = 0; r < 8; ++r) {
        float v = c1[j][r] + bias;
        v = v > 0.f ? v : 0.f;
        int l2, e2;
        a_map(r + mBase, kl2, l2, e2);
        hF[tk2 * FRAG_ELEMS + l2 * 16 + e2] = (__bf16)v;
      }
    }
  }
  __syncthreads();

  // ---- Layer 2: x0 = h @ W2 + b2; wave w owns cols [w*16, w*16+16) ----
  v8f c2 = vzero;
  for (int tk = 0; tk < KT2; ++tk) {
    v16bf a = *reinterpret_cast<const v16bf*>(&hF[tk * FRAG_ELEMS + lane * 16]);
    const v16bf bm = *reinterpret_cast<const v16bf*>(
        &w2p[((size_t)wave * KT2 + tk) * FRAG_ELEMS + lane * 16]);
    c2 = __builtin_amdgcn_wmma_f32_16x16x32_bf16(false, a, false, bm,
                                                 (short)0, c2, false, false);
  }

  // ---- Masked segment max over rows, then global float-max via u32 atomic ----
  {
    const int mBase = (lane >= 16) ? 8 : 0;
    const int n0    = wave * 16 + (lane & 15);
    const float bias = b2[n0];
    float mx = -INFINITY;
#pragma unroll
    for (int r = 0; r < 8; ++r) {
      int row = row0 + r + mBase;
      float v = c2[r] + bias;
      bool ok = (row >= start) && (row < end);
      mx = ok ? fmaxf(mx, v) : mx;
    }
    mx = fmaxf(mx, __shfl_xor(mx, 16, 32));  // combine row halves (same column)
    if (lane < 16) atomicMax(&acc[b * D_OUT + n0], enc_f32(mx));
  }
}

// Decode pooled maxes -> x1, and per-batch argmax (first-max tie-break) -> probs
__global__ void finish(const unsigned* __restrict__ acc, float* __restrict__ out) {
  int b = blockIdx.x;
  int c = threadIdx.x;  // 128
  float v = dec_f32(acc[b * D_OUT + c]);
  out[b * D_OUT + c] = v;
  __shared__ float sv[D_OUT];
  __shared__ int   si[D_OUT];
  sv[c] = v;
  si[c] = c;
  __syncthreads();
  for (int s = D_OUT / 2; s > 0; s >>= 1) {
    if (c < s) {
      float a0 = sv[c], a1 = sv[c + s];
      int   i0 = si[c], i1 = si[c + s];
      if (a1 > a0 || (a1 == a0 && i1 < i0)) { sv[c] = a1; si[c] = i1; }
    }
    __syncthreads();
  }
  if (c == 0) out[BATCH * D_OUT + b] = (float)si[0];
}

extern "C" void kernel_launch(void* const* d_in, const int* in_sizes, int n_in,
                              void* d_out, int out_size, void* d_ws, size_t ws_size,
                              hipStream_t stream) {
  const float* x          = (const float*)d_in[0];
  const float* W1         = (const float*)d_in[1];
  const float* b1         = (const float*)d_in[2];
  const float* W2         = (const float*)d_in[3];
  const float* b2         = (const float*)d_in[4];
  const int*   batch_list = (const int*)d_in[5];

  char* ws = (char*)d_ws;
  __bf16*   w1p = (__bf16*)(ws + W1P_OFF);
  __bf16*   w2p = (__bf16*)(ws + W2P_OFF);
  unsigned* acc = (unsigned*)(ws + ACC_OFF);

  init_acc<<<(BATCH * D_OUT + 255) / 256, 256, 0, stream>>>(acc);
  pack_w<<<384, 256, 0, stream>>>(W1, w1p, D_IN, D_MID);
  pack_w<<<64, 256, 0, stream>>>(W2, w2p, D_MID, D_OUT);

  dim3 grid(N_ROWS / TILE_M, BATCH);  // 128 x 64 tiles; non-segment tiles early-exit
  fused_mlp<<<grid, 256, 0, stream>>>(x, b1, b2, batch_list, w1p, w2p, acc);

  finish<<<BATCH, D_OUT, 0, stream>>>(acc, (float*)d_out);
}
